// CRFLayer_29600914604294
// MI455X (gfx1250) — matrix operand order repaired and checked by reading, not compiled, not run on hardware
//
#include <hip/hip_runtime.h>
#include <hip/hip_bf16.h>
#include <math.h>

#define BATCH 1024
#define SEQ   512
#define NT    48

typedef __attribute__((ext_vector_type(16))) _Float16 v16h;
typedef __attribute__((ext_vector_type(8)))  float    v8f;
typedef __attribute__((ext_vector_type(4)))  float    v4f;

union V16H { v16h v; unsigned u[8]; _Float16 h[16]; };

__device__ __forceinline__ unsigned pack2f16(float a, float b) {
  union { _Float16 h[2]; unsigned u; } x;
  x.h[0] = (_Float16)a; x.h[1] = (_Float16)b;
  return x.u;
}

// ---------------------------------------------------------------------------
// Kernel 1: numerator (path score). One thread per batch element; writes
// -numerator_b into ws[b].
// ---------------------------------------------------------------------------
__global__ void crf_numerator(const float* __restrict__ emis,
                              const int*   __restrict__ tags,
                              const int*   __restrict__ mask,
                              const float* __restrict__ start_trans,
                              const float* __restrict__ end_trans,
                              const float* __restrict__ trans,
                              float* __restrict__ ws_neg_numer)
{
  int b = blockIdx.x * blockDim.x + threadIdx.x;
  if (b >= BATCH) return;

  size_t base = (size_t)b * SEQ;
  int t0 = tags[base];
  float sc = start_trans[t0] + emis[base * NT + t0];
  int prev = t0;
  int last = t0;
  for (int l = 1; l < SEQ; ++l) {
    int cur = tags[base + l];
    bool keep = (mask[base + l] != 0);
    float keepf = keep ? 1.0f : 0.0f;
    sc = fmaf(keepf, trans[prev * NT + cur] + emis[(base + l) * NT + cur], sc);
    last = keep ? cur : last;
    prev = cur;                        // reference uses raw tags for prev/cur
  }
  sc += end_trans[last];
  ws_neg_numer[b] = -sc;
}

// ---------------------------------------------------------------------------
// Kernel 2: forward algorithm (normalizer) via WMMA.
// One wave per 16-batch tile.  State kept in normalized-exp form:
//   q[j] = exp(score[j] - lz),  sum_j q[j] = 1  (scalar lz per batch)
// Per step: u[j] = (Etrans^T q)[j] * exp(e[j]);  Z = sum u;  q = u/Z; lz += log Z
// -> exactly one v_log_f32 + one v_rcp_f32 per step, no max reduction.
// D layout: lane n & lane n+16 jointly hold the 48 states of batch n.
// ---------------------------------------------------------------------------
__global__ void __launch_bounds__(32)
crf_forward_wmma(const float* __restrict__ emis,
                 const int*   __restrict__ mask,
                 const float* __restrict__ start_trans,
                 const float* __restrict__ end_trans,
                 const float* __restrict__ trans,
                 float* __restrict__ denom_out)
{
  const int lane = threadIdx.x;     // 0..31
  const int n    = lane & 15;       // batch column within the 16-wide tile
  const int g    = lane >> 4;       // half-wave id
  const int b0   = blockIdx.x * 16;

  // --- A fragments: exp(trans)^T, 3 M-tiles x 2 K-blocks (K padded 48->64).
  // A k order used here (contiguous, matched by B): half h of lane -> K = 32*kb + h + 16*g.
  V16H afrag[3][2];
  #pragma unroll
  for (int m = 0; m < 3; ++m) {
    #pragma unroll
    for (int kb = 0; kb < 2; ++kb) {
      #pragma unroll
      for (int h = 0; h < 16; ++h) {
        int k = 32 * kb + h + 16 * g;
        int j = 16 * m + n;
        float v = (k < NT) ? __expf(trans[k * NT + j]) : 0.0f;
        afrag[m][kb].h[h] = (_Float16)v;
      }
    }
  }

  // --- exp(end_trans) for the 24 states this lane owns ---
  float eet[3][8];
  #pragma unroll
  for (int m = 0; m < 3; ++m)
    #pragma unroll
    for (int v = 0; v < 8; ++v)
      eet[m][v] = __expf(end_trans[16 * m + v + 8 * g]);

  // --- initial state: score0 = start_trans + emis[:,0,:] (max-normalized) ---
  const size_t row0 = (size_t)(b0 + n) * SEQ;
  float q[3][8];
  float lz;
  {
    float s0[3][8];
    float mx = -INFINITY;
    #pragma unroll
    for (int m = 0; m < 3; ++m)
      #pragma unroll
      for (int v = 0; v < 8; ++v) {
        int j = 16 * m + v + 8 * g;
        s0[m][v] = start_trans[j] + emis[row0 * NT + j];
        mx = fmaxf(mx, s0[m][v]);
      }
    mx = fmaxf(mx, __shfl_xor(mx, 16));
    float Z = 0.0f;
    #pragma unroll
    for (int m = 0; m < 3; ++m)
      #pragma unroll
      for (int v = 0; v < 8; ++v) {
        q[m][v] = __expf(s0[m][v] - mx);
        Z += q[m][v];
      }
    Z += __shfl_xor(Z, 16);
    float r = __builtin_amdgcn_rcpf(Z);
    #pragma unroll
    for (int m = 0; m < 3; ++m)
      #pragma unroll
      for (int v = 0; v < 8; ++v)
        q[m][v] *= r;
    lz = mx + __logf(Z);
  }

  // --- software-pipelined emis / mask loads (non-temporal stream) ---
  v4f e0[3], e1[3];
  {
    const float* erow = emis + (row0 + 1) * NT;
    #pragma unroll
    for (int m = 0; m < 3; ++m) {
      e0[m] = __builtin_nontemporal_load(
                reinterpret_cast<const v4f*>(erow + 16 * m + 8 * g));
      e1[m] = __builtin_nontemporal_load(
                reinterpret_cast<const v4f*>(erow + 16 * m + 8 * g + 4));
    }
  }
  int mcur = mask[row0 + 1];

  // --- sequential scan over time ---
  for (int t = 1; t < SEQ; ++t) {
    // pack q to f16 pairs and exchange halves (xor-16)
    unsigned pk[3][4];
    #pragma unroll
    for (int m = 0; m < 3; ++m)
      #pragma unroll
      for (int p = 0; p < 4; ++p)
        pk[m][p] = pack2f16(q[m][2 * p], q[m][2 * p + 1]);

    unsigned rv[3][4];
    #pragma unroll
    for (int m = 0; m < 3; ++m)
      #pragma unroll
      for (int p = 0; p < 4; ++p)
        rv[m][p] = __shfl_xor(pk[m][p], 16);

    // B fragments (32x16 f16): lanes 0-15 -> K 0-15 (+32), lanes 16-31 -> K 16-31
    V16H b1, b2;
    #pragma unroll
    for (int p = 0; p < 4; ++p) {
      b1.u[p]     = g ? rv[1][p] : pk[0][p];   // K 0..7   | 16..23
      b1.u[p + 4] = g ? pk[1][p] : rv[0][p];   // K 8..15  | 24..31
      b2.u[p]     = g ? 0u       : pk[2][p];   // K 32..39 | 48..55 (pad)
      b2.u[p + 4] = g ? 0u       : rv[2][p];   // K 40..47 | 56..63 (pad)
    }

    // issue next step's loads early, branchlessly (row index clamped in-bounds;
    // last iteration harmlessly re-reads row SEQ-1)
    const int tn = (t + 1 < SEQ) ? (t + 1) : (SEQ - 1);
    const float* enext = emis + (row0 + tn) * NT;
    v4f n0[3], n1[3];
    #pragma unroll
    for (int m = 0; m < 3; ++m) {
      n0[m] = __builtin_nontemporal_load(
                reinterpret_cast<const v4f*>(enext + 16 * m + 8 * g));
      n1[m] = __builtin_nontemporal_load(
                reinterpret_cast<const v4f*>(enext + 16 * m + 8 * g + 4));
    }
    int mnext = mask[row0 + tn];

    const bool  keep  = (mcur != 0);
    const float keepf = keep ? 1.0f : 0.0f;

    // u[j] = (Etrans^T q)[j] * exp(e[j]); accumulate Z
    float u[3][8];
    float Z = 0.0f;
    #pragma unroll
    for (int m = 0; m < 3; ++m) {
      v8f acc = {};
      acc = __builtin_amdgcn_wmma_f32_16x16x32_f16(false, afrag[m][0].v,
                                                   false, b1.v,
                                                   (short)0, acc, false, false);
      acc = __builtin_amdgcn_wmma_f32_16x16x32_f16(false, afrag[m][1].v,
                                                   false, b2.v,
                                                   (short)0, acc, false, false);
      float en[8] = {e0[m][0], e0[m][1], e0[m][2], e0[m][3],
                     e1[m][0], e1[m][1], e1[m][2], e1[m][3]};
      #pragma unroll
      for (int v = 0; v < 8; ++v) {
        u[m][v] = acc[v] * __expf(en[v]);
        Z += u[m][v];
      }
    }
    Z += __shfl_xor(Z, 16);

    // normalize; single log per step; branchless masked update
    float r = __builtin_amdgcn_rcpf(Z);
    #pragma unroll
    for (int m = 0; m < 3; ++m)
      #pragma unroll
      for (int v = 0; v < 8; ++v)
        q[m][v] = keep ? (u[m][v] * r) : q[m][v];
    lz = fmaf(keepf, __logf(Z), lz);

    // rotate pipelined loads
    #pragma unroll
    for (int m = 0; m < 3; ++m) { e0[m] = n0[m]; e1[m] = n1[m]; }
    mcur = mnext;
  }

  // --- final: denom = lz + log( sum_j q[j] * exp(end_trans[j]) ) ---
  float S = 0.0f;
  #pragma unroll
  for (int m = 0; m < 3; ++m)
    #pragma unroll
    for (int v = 0; v < 8; ++v)
      S = fmaf(q[m][v], eet[m][v], S);
  S += __shfl_xor(S, 16);

  if (g == 0)
    denom_out[b0 + n] = lz + __logf(S);
}

// ---------------------------------------------------------------------------
// Kernel 3: deterministic final reduction of 2048 partials -> out[0].
// ---------------------------------------------------------------------------
__global__ void crf_reduce(const float* __restrict__ ws, float* __restrict__ out)
{
  __shared__ float sm[256];
  float v = 0.0f;
  for (int i = threadIdx.x; i < 2 * BATCH; i += 256)
    v += ws[i];
  sm[threadIdx.x] = v;
  __syncthreads();
  for (int off = 128; off > 0; off >>= 1) {
    if (threadIdx.x < off) sm[threadIdx.x] += sm[threadIdx.x + off];
    __syncthreads();
  }
  if (threadIdx.x == 0) out[0] = sm[0];
}

// ---------------------------------------------------------------------------
extern "C" void kernel_launch(void* const* d_in, const int* in_sizes, int n_in,
                              void* d_out, int out_size, void* d_ws, size_t ws_size,
                              hipStream_t stream)
{
  (void)in_sizes; (void)n_in; (void)out_size; (void)ws_size;

  const float* emis        = (const float*)d_in[0];
  const int*   tags        = (const int*)  d_in[1];
  const int*   mask        = (const int*)  d_in[2];
  const float* start_trans = (const float*)d_in[3];
  const float* end_trans   = (const float*)d_in[4];
  const float* trans       = (const float*)d_in[5];

  float* ws  = (float*)d_ws;          // [0..1023] = -numerator, [1024..2047] = denominator
  float* out = (float*)d_out;

  crf_numerator<<<BATCH / 256, 256, 0, stream>>>(emis, tags, mask, start_trans,
                                                 end_trans, trans, ws);
  crf_forward_wmma<<<BATCH / 16, 32, 0, stream>>>(emis, mask, start_trans,
                                                  end_trans, trans, ws + BATCH);
  crf_reduce<<<1, 256, 0, stream>>>(ws, out);
}
